// BinaryConv2dSkip1x1_41609643163690
// MI455X (gfx1250) — 16-bit and 8-bit matrix operand order repaired and checked by reading, compiled, not run on hardware
//
#include <hip/hip_runtime.h>
#include <stdint.h>

typedef __attribute__((ext_vector_type(2)))  int      v2i;
typedef __attribute__((ext_vector_type(4)))  int      v4i;
typedef __attribute__((ext_vector_type(8)))  int      v8i;
typedef __attribute__((ext_vector_type(8)))  float    v8f;
typedef __attribute__((ext_vector_type(16))) _Float16 v16h;

namespace {
constexpr int CIN   = 64;
constexpr int COUT  = 64;
constexpr int IH    = 256;
constexpr int IW    = 256;
constexpr int NB    = 8;
constexpr int TROWS = 4;
constexpr int TCOLS = 32;
constexpr int SROWS = TROWS + 2;
constexpr int SCOLS = TCOLS + 2;
constexpr int SPITCH = 80;
constexpr int WPITCH = 592;
constexpr int XPITCH = 72;
constexpr int KTAPS  = 576;
}

__global__ void bq_prep_weights(const float* __restrict__ w,
                                signed char* __restrict__ bw,
                                float* __restrict__ sf) {
  int c = threadIdx.x;
  if (c >= COUT) return;
  float s = 0.f;
  for (int i = 0; i < KTAPS; ++i) s += fabsf(w[c * KTAPS + i]);
  sf[c] = s * (1.0f / 576.0f);
  for (int cin = 0; cin < CIN; ++cin)
    for (int ky = 0; ky < 3; ++ky)
      for (int kx = 0; kx < 3; ++kx) {
        float v = w[((c * CIN + cin) * 3 + ky) * 3 + kx];
        bw[c * KTAPS + (ky * 3 + kx) * 64 + cin] =
            (v > 0.f) ? 1 : ((v < 0.f) ? -1 : 0);
      }
}

__global__ __launch_bounds__(256) void bq_fused(
    const float* __restrict__ xg,  const float* __restrict__ mb,
    const signed char* __restrict__ bwg, const float* __restrict__ sfg,
    const float* __restrict__ pb0, const float* __restrict__ prw,
    const float* __restrict__ pb1, const float* __restrict__ swg,
    const float* __restrict__ sbg, float* __restrict__ outp) {
  __shared__ __align__(16) signed char sSign[SROWS * SCOLS * SPITCH];
  __shared__ __align__(16) signed char sBW[COUT * WPITCH];
  __shared__ __align__(16) _Float16    sX[TROWS * TCOLS * XPITCH];
  __shared__ __align__(16) _Float16    sSW[COUT * XPITCH];
  __shared__ float sPar[5 * 64];
  __shared__ float sMB[64];

  const int tid = threadIdx.x;
  const int w0  = blockIdx.x * TCOLS;
  const int h0  = blockIdx.y * TROWS;
  const int bz  = blockIdx.z;

  if (tid < 64) {
    sPar[tid]       = sfg[tid];
    sPar[64 + tid]  = pb0[tid];
    sPar[128 + tid] = prw[tid];
    sPar[192 + tid] = pb1[tid];
    sPar[256 + tid] = sbg[tid];
    sMB[tid]        = mb[tid];
  }
  __syncthreads();

  {
    const int* src = (const int*)bwg;
    int* dst = (int*)sBW;
    for (int i = tid; i < COUT * 144; i += 256) {
      int c = i / 144, j = i - c * 144;
      dst[c * 148 + j] = src[i];
    }
  }
  for (int i = tid; i < COUT * CIN; i += 256) {
    int c = i >> 6, k = i & 63;
    sSW[c * XPITCH + k] = (_Float16)swg[i];
  }
  for (int i = tid; i < TROWS * TCOLS * CIN; i += 256) {
    int cin = i >> 7, p = i & 127;
    int r = p >> 5, c = p & 31;
    float v = xg[((bz * CIN + cin) * IH + (h0 + r)) * IW + (w0 + c)];
    sX[p * XPITCH + cin] = (_Float16)v;
  }
  for (int i = tid; i < SROWS * CIN * SCOLS; i += 256) {
    int srow = i / (CIN * SCOLS);
    int rem  = i - srow * (CIN * SCOLS);
    int cin  = rem / SCOLS;
    int scol = rem - cin * SCOLS;
    int gh = h0 + srow - 1, gw = w0 + scol - 1;
    signed char s = 0;
    if ((unsigned)gh < (unsigned)IH && (unsigned)gw < (unsigned)IW) {
      float v = xg[((bz * CIN + cin) * IH + gh) * IW + gw] + sMB[cin];
      s = (v > 0.f) ? 1 : ((v < 0.f) ? -1 : 0);
    }
    sSign[(srow * SCOLS + scol) * SPITCH + cin] = s;
  }
  __syncthreads();

  const int  lane   = tid & 31;
  const int  wid    = tid >> 5;
  const int  cbase  = (wid & 3) * 16;
  const int  rh     = wid >> 2;
  const int  m      = lane & 15;
  const bool hihalf = lane >= 16;

  v8i accI[4] = {};
  v8f accF[4] = {};

  union UI8  { v2i d2[4]; v4i d4[2]; v8i v; };
  union UH16 { v4i d4[2]; v16h v; };

  const int aBase = (cbase + m) * WPITCH + (hihalf ? 16 : 0);
  int pixBase[4];
#pragma unroll
  for (int nt = 0; nt < 4; ++nt) {
    const int rl = rh * 2 + (nt >> 1);
    pixBase[nt] = (rl * SCOLS + (nt & 1) * 16 + m) * SPITCH + (hihalf ? 16 : 0);
  }

#pragma unroll
  for (int tap = 0; tap < 9; ++tap) {
    const int dy = tap / 3;
    const int dx = tap - 3 * dy;
    const int tapOff = (dy * SCOLS + dx) * SPITCH;
    UI8 a;
    {
      const v4i* q = (const v4i*)(sBW + aBase + tap * 64);
      a.d4[0] = q[0]; a.d4[1] = q[2];
    }
#pragma unroll
    for (int nt = 0; nt < 4; ++nt) {
      UI8 b;
      const v4i* q = (const v4i*)(sSign + pixBase[nt] + tapOff);
      b.d4[0] = q[0]; b.d4[1] = q[2];
      accI[nt] = __builtin_amdgcn_wmma_i32_16x16x64_iu8(
          true, a.v, true, b.v, accI[nt], false, false);
    }
  }

#pragma unroll
  for (int ks = 0; ks < 2; ++ks) {
    UH16 a;
    {
      const v4i* q = (const v4i*)((const signed char*)sSW +
          ((cbase + m) * XPITCH + ks * 32 + (hihalf ? 8 : 0)) * 2);
      a.d4[0] = q[0]; a.d4[1] = q[2];
    }
#pragma unroll
    for (int nt = 0; nt < 4; ++nt) {
      const int rl = rh * 2 + (nt >> 1);
      const int p  = rl * TCOLS + (nt & 1) * 16 + m;
      UH16 b;
      const v4i* q = (const v4i*)((const signed char*)sX +
          (p * XPITCH + ks * 32 + (hihalf ? 8 : 0)) * 2);
      b.d4[0] = q[0]; b.d4[1] = q[2];
      accF[nt] = __builtin_amdgcn_wmma_f32_16x16x32_f16(
          false, a.v, false, b.v, (short)0, accF[nt], false, false);
    }
  }

#pragma unroll
  for (int r = 0; r < 8; ++r) {
    const int cout = cbase + r + (hihalf ? 8 : 0);
    const float sfc = sPar[cout];
    const float b0  = sPar[64 + cout];
    const float aw  = sPar[128 + cout];
    const float b1  = sPar[192 + cout];
    const float sbv = sPar[256 + cout];
#pragma unroll
    for (int nt = 0; nt < 4; ++nt) {
      float f = (float)accI[nt][r] * sfc + b0;
      f = fmaxf(f, 0.f) + aw * fminf(f, 0.f) + b1;
      const float o = f + accF[nt][r] + sbv;
      const int rl = rh * 2 + (nt >> 1);
      const int wg = w0 + (nt & 1) * 16 + m;
      outp[((bz * COUT + cout) * IH + (h0 + rl)) * IW + wg] = o;
    }
  }
}

extern "C" void kernel_launch(void* const* d_in, const int* in_sizes, int n_in,
                              void* d_out, int out_size, void* d_ws, size_t ws_size,
                              hipStream_t stream) {
  const float* x   = (const float*)d_in[0];
  const float* mbp = (const float*)d_in[1];
  const float* wgt = (const float*)d_in[2];
  const float* pb0 = (const float*)d_in[3];
  const float* prw = (const float*)d_in[4];
  const float* pb1 = (const float*)d_in[5];
  const float* sw  = (const float*)d_in[6];
  const float* sb  = (const float*)d_in[7];
  float* out = (float*)d_out;

  signed char* bw = (signed char*)d_ws;
  float* sf = (float*)((char*)d_ws + COUT * KTAPS);

  bq_prep_weights<<<1, 64, 0, stream>>>(wgt, bw, sf);
  dim3 grid(IW / TCOLS, IH / TROWS, NB);
  bq_fused<<<grid, 256, 0, stream>>>(x, mbp, bw, sf, pb0, prw, pb1, sw, sb, out);
}
